// CausalSelfAttention_64020782514268
// MI455X (gfx1250) — matrix operand order repaired and checked by reading, not compiled, not run on hardware
//
#include <hip/hip_runtime.h>
#include <hip/hip_bf16.h>

// ---------------- problem constants ----------------
#define C_EMBD 768
#define NHEAD  12
#define DHEAD  64
#define SEQ    4096
#define BATCH  2
#define MTOT   (BATCH * SEQ)   // 8192 tokens
#define N3     (3 * C_EMBD)    // 2304

typedef __bf16 bf16;
typedef __attribute__((ext_vector_type(16))) __bf16 v16bf;
typedef __attribute__((ext_vector_type(8)))  float  v8f;

#define WMMA_BF16(A, B, Cc) \
  __builtin_amdgcn_wmma_f32_16x16x32_bf16(false, (A), false, (B), (short)0, (Cc), false, false)

// ---- gfx1250 async global->LDS path (probe-guarded) ----
#if defined(__has_builtin)
#if __has_builtin(__builtin_amdgcn_global_load_async_to_lds_b128) && \
    __has_builtin(__builtin_amdgcn_s_wait_asynccnt)
#define HAVE_ASYNC 1
#endif
#endif
#ifndef HAVE_ASYNC
#define HAVE_ASYNC 0
#endif

// Builtin signature (from hipcc diagnostic): parameters are pointers to
// 'int __attribute__((vector_size(16)))' in addrspace(1) (src) / addrspace(3) (dst).
typedef int v4i __attribute__((vector_size(16)));
typedef __attribute__((address_space(1))) v4i as1_v4i;
typedef __attribute__((address_space(3))) v4i as3_v4i;

static __device__ __forceinline__ void stage_b128(void* dst_lds, const void* src_glob) {
#if HAVE_ASYNC
  __builtin_amdgcn_global_load_async_to_lds_b128((as1_v4i*)src_glob, (as3_v4i*)dst_lds,
                                                 0, 0);
#else
  *(uint4*)dst_lds = *(const uint4*)src_glob;
#endif
}
static __device__ __forceinline__ void wait_async_le2() {
#if HAVE_ASYNC
  __builtin_amdgcn_s_wait_asynccnt(2);
#endif
}
static __device__ __forceinline__ void wait_async_le0() {
#if HAVE_ASYNC
  __builtin_amdgcn_s_wait_asynccnt(0);
#endif
}

// -------- fragment loaders (match CDNA5 ISA 16-bit A/B VGPR layouts) --------
// A (16x32, MxK): lane<16 holds row=lane, K [0..7] and [16..23];
//                 lane>=16 holds row=lane-16, K [8..15] and [24..31].
static __device__ __forceinline__ v16bf load_a_frag(const bf16* base, int ld, int lane) {
  const int r  = lane & 15;
  const int c0 = (lane < 16) ? 0 : 8;
  const bf16* p = base + (size_t)r * ld + c0;
  union { uint4 q[2]; v16bf v; } u;
  u.q[0] = *(const uint4*)(p);
  u.q[1] = *(const uint4*)(p + 16);
  return u.v;
}
// B (32x16, KxN), source stored column-major (ld = stride between columns):
// lane<16 holds col=lane, K [0..7] and [16..23]; lane>=16 holds col=lane-16, K [8..15] and [24..31].
static __device__ __forceinline__ v16bf load_b_frag(const bf16* base, int ld, int lane) {
  const int c  = lane & 15;
  const int k0 = (lane < 16) ? 0 : 8;
  const bf16* p = base + (size_t)c * ld + k0;
  union { uint4 q[2]; v16bf v; } u;
  u.q[0] = *(const uint4*)(p);
  u.q[1] = *(const uint4*)(p + 16);
  return u.v;
}

// ---------------- conversion kernels ----------------
__global__ void cvt_f32_to_bf16(const float* __restrict__ x, bf16* __restrict__ xb, int n) {
  int i = blockIdx.x * blockDim.x + threadIdx.x;
  if (i < n) xb[i] = (bf16)x[i];
}
// W stored [K][N] row-major in fp32 -> WT stored [N][K] bf16 (column-major of W).
__global__ void cvt_w_transpose(const float* __restrict__ w, bf16* __restrict__ wT,
                                int K, int N) {
  int i = blockIdx.x * blockDim.x + threadIdx.x;
  if (i < K * N) {
    int k = i / N, n = i % N;
    wT[(size_t)n * K + k] = (bf16)w[i];
  }
}

// =================== shared-B GEMM core ===================
// Block: 128 threads = 4 waves. Block tile: 128(M) x 64(N). Wave tile: 32 x 64.
// B panel (64 cols x 32 K, bf16, column-major) double-buffered in LDS, filled
// with async global->LDS copies one K-step ahead of the WMMAs.
// LDS layout per buffer: 64 rows (columns n) x stride 40 bf16 (80 B, 16B-aligned).
#define BSTRIDE 40

// Fills Bst[buf] with WT[n0+0..63][k0..k0+31]. tid in [0,128).
static __device__ __forceinline__ void stage_b_panel(bf16* buf, const bf16* WT_n0,
                                                     int k0, int tid) {
  const int n = tid >> 1;            // 0..63
  const int c = (tid & 1) * 2;       // chunk base {0,2}, chunks of 8 bf16
  const bf16* src = WT_n0 + (size_t)n * C_EMBD + k0 + c * 8;
  bf16* dst = buf + n * BSTRIDE + c * 8;
  stage_b128(dst, src);
  stage_b128(dst + 8, src + 8);
}

// ---------------- QKV projection GEMM ----------------
// M=8192, N=2304, K=768. Writes Q,K as [b,h,tok,64] bf16 and V as [b,h,64,tok].
__global__ __launch_bounds__(128) void qkv_gemm(const bf16* __restrict__ Xb,
                                                const bf16* __restrict__ WaT,
                                                const float* __restrict__ bias,
                                                bf16* __restrict__ Qh,
                                                bf16* __restrict__ Kh,
                                                bf16* __restrict__ Vt) {
  __shared__ alignas(16) bf16 Bst[2][64 * BSTRIDE];
  const int tid  = threadIdx.x;
  const int lane = tid & 31;
  const int wave = tid >> 5;
  const int m0 = blockIdx.x * 128 + wave * 32;
  const int n0 = blockIdx.y * 64;
  const bf16* WT_n0 = WaT + (size_t)n0 * C_EMBD;
  const bf16* arow0 = Xb + (size_t)m0 * C_EMBD;
  const bf16* arow1 = arow0 + (size_t)16 * C_EMBD;

  v8f acc[2][4] = {{v8f{}, v8f{}, v8f{}, v8f{}}, {v8f{}, v8f{}, v8f{}, v8f{}}};

  stage_b_panel(Bst[0], WT_n0, 0, tid);

  for (int k0 = 0; k0 < C_EMBD; k0 += 32) {
    const int cur = (k0 >> 5) & 1;
    const bool more = (k0 + 32 < C_EMBD);
    if (more) stage_b_panel(Bst[cur ^ 1], WT_n0, k0 + 32, tid);
    if (more) wait_async_le2(); else wait_async_le0();
    __syncthreads();  // Bst[cur] complete and visible block-wide

    __builtin_prefetch((const void*)(arow0 + k0 + 64), 0, 1);
    v16bf a0 = load_a_frag(arow0 + k0, C_EMBD, lane);
    v16bf a1 = load_a_frag(arow1 + k0, C_EMBD, lane);
#pragma unroll
    for (int t = 0; t < 4; ++t) {
      v16bf bfr = load_b_frag(&Bst[cur][t * 16 * BSTRIDE], BSTRIDE, lane);
      acc[0][t] = WMMA_BF16(a0, bfr, acc[0][t]);
      acc[1][t] = WMMA_BF16(a1, bfr, acc[1][t]);
    }
    __syncthreads();  // reads done before Bst[cur] is overwritten next round
  }

  const int hl = lane >> 4;
  const int ln = lane & 15;
#pragma unroll
  for (int mi = 0; mi < 2; ++mi) {
#pragma unroll
    for (int t = 0; t < 4; ++t) {
      const int n = n0 + t * 16 + ln;
      const float bs = bias[n];
#pragma unroll
      for (int j = 0; j < 8; ++j) {
        const int m = m0 + mi * 16 + j + (hl << 3);
        const float v = acc[mi][t][j] + bs;
        const bf16 bv = (bf16)v;
        const int b = m >> 12;      // / SEQ
        const int tok = m & 4095;   // % SEQ
        if (n < C_EMBD) {
          const int h = n >> 6, dh = n & 63;
          Qh[((size_t)(b * NHEAD + h) * SEQ + tok) * DHEAD + dh] = bv;
        } else if (n < 2 * C_EMBD) {
          const int n2 = n - C_EMBD, h = n2 >> 6, dh = n2 & 63;
          Kh[((size_t)(b * NHEAD + h) * SEQ + tok) * DHEAD + dh] = bv;
        } else {
          const int n2 = n - 2 * C_EMBD, h = n2 >> 6, dh = n2 & 63;
          Vt[((size_t)(b * NHEAD + h) * DHEAD + dh) * SEQ + tok] = bv;
        }
      }
    }
  }
}

// ---------------- flash attention (causal) ----------------
// One wave owns 16 query rows of one (b,h); loops K/V in steps of 32 tokens.
__global__ __launch_bounds__(128) void attn_kernel(const bf16* __restrict__ Qh,
                                                   const bf16* __restrict__ Kh,
                                                   const bf16* __restrict__ Vt,
                                                   bf16* __restrict__ Y) {
  __shared__ alignas(16) bf16 pbuf[4][16 * 40];  // per-wave P transpose buffer
  const int lane = threadIdx.x & 31;
  const int wave = threadIdx.x >> 5;
  const int hl = lane >> 4;
  const int ln = lane & 15;

  const int bh = blockIdx.y;
  const int b = bh / NHEAD, h = bh % NHEAD;
  const bf16* Qp = Qh + (size_t)bh * SEQ * DHEAD;
  const bf16* Kp = Kh + (size_t)bh * SEQ * DHEAD;
  const bf16* Vp = Vt + (size_t)bh * DHEAD * SEQ;
  const int q0 = (blockIdx.x * 4 + wave) * 16;
  bf16* pl = pbuf[wave];

  const v16bf qa0 = load_a_frag(Qp + (size_t)q0 * DHEAD + 0, DHEAD, lane);
  const v16bf qa1 = load_a_frag(Qp + (size_t)q0 * DHEAD + 32, DHEAD, lane);

  v8f o0 = {}, o1 = {}, o2 = {}, o3 = {};
  float mrow[8], lrow[8];
#pragma unroll
  for (int j = 0; j < 8; ++j) { mrow[j] = -3.0e38f; lrow[j] = 0.0f; }

  for (int k0 = 0; k0 < q0 + 16; k0 += 32) {
    // ---- S = Q @ K^T (16 x 32 of this step) ----
    v8f s0 = {}, s1 = {};
    {
      v16bf kb = load_b_frag(Kp + (size_t)k0 * DHEAD + 0, DHEAD, lane);
      s0 = WMMA_BF16(qa0, kb, s0);
      kb = load_b_frag(Kp + (size_t)k0 * DHEAD + 32, DHEAD, lane);
      s0 = WMMA_BF16(qa1, kb, s0);
      kb = load_b_frag(Kp + (size_t)(k0 + 16) * DHEAD + 0, DHEAD, lane);
      s1 = WMMA_BF16(qa0, kb, s1);
      kb = load_b_frag(Kp + (size_t)(k0 + 16) * DHEAD + 32, DHEAD, lane);
      s1 = WMMA_BF16(qa1, kb, s1);
    }

    const float scale = 0.125f;  // 1/sqrt(64)
    const int col0 = k0 + ln;
    const int col1 = k0 + 16 + ln;
    const bool need_mask = (k0 + 31 > q0);

#pragma unroll
    for (int j = 0; j < 8; ++j) {
      const int row = q0 + j + (hl << 3);
      float v0 = s0[j] * scale;
      float v1 = s1[j] * scale;
      if (need_mask) {
        if (col0 > row) v0 = -3.0e38f;
        if (col1 > row) v1 = -3.0e38f;
      }
      float cmax = fmaxf(v0, v1);
      cmax = fmaxf(cmax, __shfl_xor(cmax, 1, 32));
      cmax = fmaxf(cmax, __shfl_xor(cmax, 2, 32));
      cmax = fmaxf(cmax, __shfl_xor(cmax, 4, 32));
      cmax = fmaxf(cmax, __shfl_xor(cmax, 8, 32));
      const float mnew = fmaxf(mrow[j], cmax);
      const float corr = __expf(mrow[j] - mnew);
      const float p0 = __expf(v0 - mnew);
      const float p1 = __expf(v1 - mnew);
      float rs = p0 + p1;
      rs += __shfl_xor(rs, 1, 32);
      rs += __shfl_xor(rs, 2, 32);
      rs += __shfl_xor(rs, 4, 32);
      rs += __shfl_xor(rs, 8, 32);
      lrow[j] = lrow[j] * corr + rs;
      mrow[j] = mnew;
      o0[j] *= corr; o1[j] *= corr; o2[j] *= corr; o3[j] *= corr;
      const int r = j + (hl << 3);
      pl[r * 40 + ln] = (bf16)p0;
      pl[r * 40 + 16 + ln] = (bf16)p1;
    }

    __threadfence_block();  // order DS stores before DS loads (per-wave, in-order LDS)

    const v16bf pa = load_a_frag(pl, 40, lane);
    v16bf vb;
    vb = load_b_frag(Vp + (size_t)(0 * 16) * SEQ + k0, SEQ, lane);
    o0 = WMMA_BF16(pa, vb, o0);
    vb = load_b_frag(Vp + (size_t)(1 * 16) * SEQ + k0, SEQ, lane);
    o1 = WMMA_BF16(pa, vb, o1);
    vb = load_b_frag(Vp + (size_t)(2 * 16) * SEQ + k0, SEQ, lane);
    o2 = WMMA_BF16(pa, vb, o2);
    vb = load_b_frag(Vp + (size_t)(3 * 16) * SEQ + k0, SEQ, lane);
    o3 = WMMA_BF16(pa, vb, o3);

    __threadfence_block();  // protect LDS WAR before next iteration's stores
  }

  const size_t ybase = ((size_t)b * SEQ) * C_EMBD + (size_t)h * DHEAD;
#pragma unroll
  for (int j = 0; j < 8; ++j) {
    const int tok = q0 + j + (hl << 3);
    const float inv = 1.0f / lrow[j];
    const size_t rb = ybase + (size_t)tok * C_EMBD + ln;
    Y[rb + 0]  = (bf16)(o0[j] * inv);
    Y[rb + 16] = (bf16)(o1[j] * inv);
    Y[rb + 32] = (bf16)(o2[j] * inv);
    Y[rb + 48] = (bf16)(o3[j] * inv);
  }
}

// ---------------- output projection GEMM ----------------
// out[m,n] = Y[m,:] @ W_out[:,n] + b_out[n], fp32 output. Same shared-B scheme.
__global__ __launch_bounds__(128) void out_proj(const bf16* __restrict__ Y,
                                                const bf16* __restrict__ WoT,
                                                const float* __restrict__ bias,
                                                float* __restrict__ out) {
  __shared__ alignas(16) bf16 Bst[2][64 * BSTRIDE];
  const int tid  = threadIdx.x;
  const int lane = tid & 31;
  const int wave = tid >> 5;
  const int m0 = blockIdx.x * 128 + wave * 32;
  const int n0 = blockIdx.y * 64;
  const bf16* WT_n0 = WoT + (size_t)n0 * C_EMBD;
  const bf16* arow0 = Y + (size_t)m0 * C_EMBD;
  const bf16* arow1 = arow0 + (size_t)16 * C_EMBD;

  v8f acc[2][4] = {{v8f{}, v8f{}, v8f{}, v8f{}}, {v8f{}, v8f{}, v8f{}, v8f{}}};

  stage_b_panel(Bst[0], WT_n0, 0, tid);

  for (int k0 = 0; k0 < C_EMBD; k0 += 32) {
    const int cur = (k0 >> 5) & 1;
    const bool more = (k0 + 32 < C_EMBD);
    if (more) stage_b_panel(Bst[cur ^ 1], WT_n0, k0 + 32, tid);
    if (more) wait_async_le2(); else wait_async_le0();
    __syncthreads();

    __builtin_prefetch((const void*)(arow0 + k0 + 64), 0, 1);
    v16bf a0 = load_a_frag(arow0 + k0, C_EMBD, lane);
    v16bf a1 = load_a_frag(arow1 + k0, C_EMBD, lane);
#pragma unroll
    for (int t = 0; t < 4; ++t) {
      v16bf bfr = load_b_frag(&Bst[cur][t * 16 * BSTRIDE], BSTRIDE, lane);
      acc[0][t] = WMMA_BF16(a0, bfr, acc[0][t]);
      acc[1][t] = WMMA_BF16(a1, bfr, acc[1][t]);
    }
    __syncthreads();
  }

  const int hl = lane >> 4;
  const int ln = lane & 15;
#pragma unroll
  for (int mi = 0; mi < 2; ++mi) {
#pragma unroll
    for (int t = 0; t < 4; ++t) {
      const int n = n0 + t * 16 + ln;
      const float bs = bias[n];
#pragma unroll
      for (int j = 0; j < 8; ++j) {
        const int m = m0 + mi * 16 + j + (hl << 3);
        out[(size_t)m * C_EMBD + n] = acc[mi][t][j] + bs;
      }
    }
  }
}

// ---------------- host-side launch ----------------
extern "C" void kernel_launch(void* const* d_in, const int* in_sizes, int n_in,
                              void* d_out, int out_size, void* d_ws, size_t ws_size,
                              hipStream_t stream) {
  const float* x  = (const float*)d_in[0];
  const float* Wa = (const float*)d_in[1];
  const float* ba = (const float*)d_in[2];
  const float* Wo = (const float*)d_in[3];
  const float* bo = (const float*)d_in[4];
  float* out = (float*)d_out;

  char* ws = (char*)d_ws;
  const size_t SZ_XB  = (size_t)MTOT * C_EMBD * sizeof(bf16);               // 12.6 MB
  const size_t SZ_WAT = (size_t)C_EMBD * N3 * sizeof(bf16);                 //  3.5 MB
  const size_t SZ_WOT = (size_t)C_EMBD * C_EMBD * sizeof(bf16);             //  1.2 MB
  const size_t SZ_H   = (size_t)BATCH * NHEAD * SEQ * DHEAD * sizeof(bf16); // 12.6 MB

  size_t off = 0;
  bf16* Xb  = (bf16*)(ws + off); off += SZ_XB;
  bf16* WaT = (bf16*)(ws + off); off += SZ_WAT;
  bf16* WoT = (bf16*)(ws + off); off += SZ_WOT;
  bf16* Qh  = (bf16*)(ws + off); off += SZ_H;
  bf16* Kh  = (bf16*)(ws + off); off += SZ_H;
  bf16* Vt  = (bf16*)(ws + off); off += SZ_H;
  bf16* Yb  = (bf16*)(ws + off); off += SZ_XB;
  (void)ws_size; (void)in_sizes; (void)n_in; (void)out_size;

  const int nx = MTOT * C_EMBD;
  cvt_f32_to_bf16<<<(nx + 255) / 256, 256, 0, stream>>>(x, Xb, nx);
  cvt_w_transpose<<<(C_EMBD * N3 + 255) / 256, 256, 0, stream>>>(Wa, WaT, C_EMBD, N3);
  cvt_w_transpose<<<(C_EMBD * C_EMBD + 255) / 256, 256, 0, stream>>>(Wo, WoT, C_EMBD, C_EMBD);

  qkv_gemm<<<dim3(MTOT / 128, N3 / 64), 128, 0, stream>>>(Xb, WaT, ba, Qh, Kh, Vt);
  attn_kernel<<<dim3(SEQ / 64, BATCH * NHEAD), 128, 0, stream>>>(Qh, Kh, Vt, Yb);
  out_proj<<<dim3(MTOT / 128, C_EMBD / 64), 128, 0, stream>>>(Yb, WoT, bo, out);
}
